// TransformerBlock_30021821399763
// MI455X (gfx1250) — matrix operand order repaired and checked by reading, not compiled, not run on hardware
//
#include <hip/hip_runtime.h>
#include <hip/hip_bf16.h>
#include <math.h>

typedef _Float16 half_t;
typedef __attribute__((ext_vector_type(16))) _Float16 v16h;
typedef __attribute__((ext_vector_type(4)))  _Float16 v4h;
typedef __attribute__((ext_vector_type(8)))  float    v8f;
typedef __attribute__((ext_vector_type(4)))  unsigned int u32x4;
typedef __attribute__((ext_vector_type(8)))  int i32x8;
typedef __attribute__((ext_vector_type(4)))  int i32x4;

#define LDS_PAD 8

// ---------------- f32 -> f16 convert (vectorized, n % 4 == 0) ----------------
__global__ void cvt_f32_f16(const float* __restrict__ in, half_t* __restrict__ out, int n) {
  int i = (blockIdx.x * blockDim.x + threadIdx.x) * 4;
  const int stride = gridDim.x * blockDim.x * 4;
  for (; i < n; i += stride) {
    const float4 v = *(const float4*)(in + i);
    v4h h;
    h.x = (half_t)v.x; h.y = (half_t)v.y; h.z = (half_t)v.z; h.w = (half_t)v.w;
    *(v4h*)(out + i) = h;
  }
}

// ---------------- LayerNorm: fp32 in -> f16 out (H % 4 == 0) ----------------
__global__ __launch_bounds__(256) void layernorm_f16(
    const float* __restrict__ X, const float* __restrict__ gamma,
    const float* __restrict__ beta, half_t* __restrict__ Y, int H) {
  const int row = blockIdx.x;
  const float* x = X + (size_t)row * H;
  float s = 0.f, s2 = 0.f;
  for (int i = threadIdx.x * 4; i < H; i += 1024) {
    const float4 v = *(const float4*)(x + i);
    s  += v.x + v.y + v.z + v.w;
    s2 += v.x * v.x + v.y * v.y + v.z * v.z + v.w * v.w;
  }
  #pragma unroll
  for (int off = 16; off > 0; off >>= 1) {
    s  += __shfl_xor(s,  off, 32);
    s2 += __shfl_xor(s2, off, 32);
  }
  __shared__ float red[2][8];
  const int wave = threadIdx.x >> 5;
  if ((threadIdx.x & 31) == 0) { red[0][wave] = s; red[1][wave] = s2; }
  __syncthreads();
  float ts = 0.f, ts2 = 0.f;
  #pragma unroll
  for (int w = 0; w < 8; ++w) { ts += red[0][w]; ts2 += red[1][w]; }
  const float mean = ts / H;
  const float rstd = rsqrtf(ts2 / H - mean * mean + 1e-5f);
  half_t* y = Y + (size_t)row * H;
  for (int i = threadIdx.x * 4; i < H; i += 1024) {
    const float4 v  = *(const float4*)(x + i);
    const float4 g4 = *(const float4*)(gamma + i);
    const float4 b4 = *(const float4*)(beta + i);
    v4h h;
    h.x = (half_t)((v.x - mean) * rstd * g4.x + b4.x);
    h.y = (half_t)((v.y - mean) * rstd * g4.y + b4.y);
    h.z = (half_t)((v.z - mean) * rstd * g4.z + b4.z);
    h.w = (half_t)((v.w - mean) * rstd * g4.w + b4.w);
    *(v4h*)(y + i) = h;
  }
}

// helper: assemble v16h from two 16B LDS chunks
__device__ __forceinline__ v16h frag16(const half_t* p0, const half_t* p1) {
  v16h r;
  *(uint4*)&r       = *(const uint4*)p0;
  *((uint4*)&r + 1) = *(const uint4*)p1;
  return r;
}

// ---- TDM: async 2D tile (rows x 32 f16) global -> LDS, padded to 80B row stride ----
// Zero-fills rows past rows_avail (hardware OOB handling). 6-arg clang-23 builtin form.
__device__ __forceinline__ void tdm_load_a_tile(const half_t* gaddr, unsigned lds_off,
                                                unsigned rows_avail, unsigned k_avail, unsigned K) {
  const unsigned long long ga = (unsigned long long)(uintptr_t)gaddr;
  u32x4 g0;
  g0.x = 0x1u;                                   // count=1 valid descriptor
  g0.y = lds_off;                                // LDS byte address
  g0.z = (unsigned)ga;                           // global_addr[31:0]
  g0.w = (unsigned)(ga >> 32) | 0x80000000u;     // global_addr[56:32] | type=2 ("image")
  i32x8 g1;
  g1[0] = (1 << 16) | (1 << 20) | (3 << 22) | (3 << 25); // data_size=2B, pad_en, every 8x8B, pad 4 dwords
  g1[1] = (int)((k_avail & 0xffffu) << 16);              // tensor_dim0[15:0]
  g1[2] = (int)((k_avail >> 16) | ((rows_avail & 0xffffu) << 16)); // dim0 hi | dim1 lo
  g1[3] = (int)((rows_avail >> 16) | (32u << 16));       // dim1 hi | tile_dim0=32
  g1[4] = 128;                                           // tile_dim1=128, tile_dim2=0
  g1[5] = (int)K;                                        // tensor_dim0_stride (elements)
  g1[6] = 0;
  g1[7] = 0;
  const i32x4 z4 = {0, 0, 0, 0};
  const i32x8 z8 = {0, 0, 0, 0, 0, 0, 0, 0};
  __builtin_amdgcn_tensor_load_to_lds(g0, g1, z4, z4, z8, 0);
}

// ---------------- WMMA GEMM: C[M,N] = A[M,K](f16) * B[K,N](f16) ----------------
// Block tile 128x128, BK=32, 8 waves (4x2), each wave -> 32x64 (8 WMMA tiles).
// A tiles arrive via double-buffered TDM (tensor_load_to_lds); B staged transposed by VALU.
template<bool HBIAS, bool HGELU, bool HRES, bool OUTH>
__global__ __launch_bounds__(256) void gemm_wmma_f16(
    const half_t* __restrict__ A, const half_t* __restrict__ Bw,
    const float* __restrict__ bias, const float* __restrict__ resid,
    float* __restrict__ Cf, half_t* __restrict__ Ch,
    int M, int N, int K) {
  constexpr int BM = 128, BN = 128, BK = 32;
  constexpr int AROW   = BK + LDS_PAD;            // 40 halves = 80B (matches TDM pad cfg)
  constexpr unsigned ABUF = BM * AROW * 2;        // 10240 B per A buffer
  __shared__ alignas(16) char lds_pool[2 * ABUF + BN * AROW * 2];
  half_t (*Bt)[AROW] = (half_t(*)[AROW])(lds_pool + 2 * ABUF);   // B tile transposed: Bt[n][k]
  const int tid  = threadIdx.x;
  const int lane = tid & 31;
  const int wave = tid >> 5;
  const int wm = wave >> 1, wn = wave & 1;
  const int l16 = lane & 15, lh = lane >> 4;
  const int bm0 = blockIdx.y * BM;
  const int bn0 = blockIdx.x * BN;
  const unsigned rows_avail = (unsigned)(M - bm0);

  v8f acc[2][4] = {};

  const int nk = K / BK;
  if (tid == 0)  // kick off first A tile (EXEC-independent DMA, one issue per block)
    tdm_load_a_tile(A + (size_t)bm0 * K, 0u, rows_avail, (unsigned)K, (unsigned)K);

  for (int kt = 0; kt < nk; ++kt) {
    const int k0 = kt * BK;
    // stage B transposed: 32x128 halves = 512 16B-chunks, 2 per thread
    #pragma unroll
    for (int c = 0; c < 2; ++c) {
      const int chunk = tid * 2 + c;
      const int r  = chunk >> 4;           // k row 0..31
      const int nc = (chunk & 15) * 8;     // n col 0..120
      const uint4 v = *(const uint4*)(Bw + (size_t)(k0 + r) * N + bn0 + nc);
      const half_t* p = (const half_t*)&v;
      #pragma unroll
      for (int e = 0; e < 8; ++e) Bt[nc + e][r] = p[e];
    }
    if (k0 + BK < K) __builtin_prefetch(Bw + (size_t)(k0 + BK + (tid >> 4)) * N + bn0, 0, 3);
    if (wave == 0) __builtin_amdgcn_s_wait_tensorcnt(0);  // issuer wave drains the DMA
    __syncthreads();                                       // publish A tile + B tile
    // overlap: issue next A tile into the other buffer while this one is consumed
    if (tid == 0 && kt + 1 < nk)
      tdm_load_a_tile(A + (size_t)bm0 * K + (k0 + BK), ((unsigned)(kt + 1) & 1u) * ABUF,
                      rows_avail, (unsigned)(K - k0 - BK), (unsigned)K);

    const half_t (*As)[AROW] = (const half_t(*)[AROW])(lds_pool + (unsigned)(kt & 1) * ABUF);
    v16h af[2], bf[4];
    #pragma unroll
    for (int i = 0; i < 2; ++i) {
      const int m = wm * 32 + i * 16 + l16;               // CDNA5 A layout: two 8-half runs
      af[i] = frag16(&As[m][lh * 8], &As[m][16 + lh * 8]);
    }
    #pragma unroll
    for (int j = 0; j < 4; ++j) {
      const int n = wn * 64 + j * 16 + l16;               // CDNA5 B layout: two 8-half runs (transposed)
      bf[j] = frag16(&Bt[n][lh * 8], &Bt[n][16 + lh * 8]);
    }
    #pragma unroll
    for (int i = 0; i < 2; ++i)
      #pragma unroll
      for (int j = 0; j < 4; ++j)
        acc[i][j] = __builtin_amdgcn_wmma_f32_16x16x32_f16(
            false, af[i], false, bf[j], (short)0, acc[i][j], false, false);
    __syncthreads();
  }

  // epilogue: C layout row = vgpr + 8*lh, col = l16
  #pragma unroll
  for (int i = 0; i < 2; ++i)
    #pragma unroll
    for (int j = 0; j < 4; ++j) {
      const int col = bn0 + wn * 64 + j * 16 + l16;
      const int row0 = bm0 + wm * 32 + i * 16 + 8 * lh;
      const float bv = HBIAS ? bias[col] : 0.f;
      #pragma unroll
      for (int v = 0; v < 8; ++v) {
        const int row = row0 + v;
        if (row >= M) continue;
        float val = acc[i][j][v] + bv;
        if (HGELU) val = 0.5f * val * (1.0f + erff(val * 0.70710678118654752f));
        if (HRES)  val += resid[(size_t)row * N + col];
        if (OUTH)  Ch[(size_t)row * N + col] = (half_t)val;
        else       Cf[(size_t)row * N + col] = val;
      }
    }
}

// ---------------- Flash attention with WMMA (hd=64, wave handles 16 q-rows) ----------------
__global__ __launch_bounds__(128) void attn_flash_wmma(
    const half_t* __restrict__ Q, const half_t* __restrict__ Km,
    const half_t* __restrict__ V, half_t* __restrict__ O,
    int B, int T, int S, int NH) {
  const int E = NH * 64;
  __shared__ alignas(16) half_t Ks[32][64 + LDS_PAD];   // K tile, row-major [s_local][d]
  __shared__ alignas(16) half_t Vt[64][32 + LDS_PAD];   // V tile transposed [d][s_local]
  __shared__ alignas(16) half_t Pl[4][16][32 + LDS_PAD];
  const int tid  = threadIdx.x;
  const int lane = tid & 31;
  const int wave = tid >> 5;
  const int l16 = lane & 15, lh = lane >> 4;
  const int h = blockIdx.y % NH, b = blockIdx.y / NH;
  const int q0 = blockIdx.x * 64 + wave * 16;

  // Q fragments (A layout): 4 x b128 loads per lane, 1/sqrt(64) folded in
  v16h aq[2];
  const half_t* qptr = Q + (size_t)(b * T + q0 + l16) * E + h * 64;
  #pragma unroll
  for (int f = 0; f < 2; ++f)
    #pragma unroll
    for (int g = 0; g < 2; ++g) {
      uint4 u = *(const uint4*)(qptr + f * 32 + lh * 16 + g * 8);
      const half_t* hp = (const half_t*)&u;
      #pragma unroll
      for (int e = 0; e < 8; ++e) aq[f][g * 8 + e] = (half_t)((float)hp[e] * 0.125f);
    }

  v16h bones;
  #pragma unroll
  for (int e = 0; e < 16; ++e) bones[e] = (half_t)1.f;

  v8f o[4] = {};
  v8f rlacc = {};          // running softmax denominator, via WMMA against ones
  float rm = -1e30f;       // running TILE max (same constant per row: exact softmax)

  for (int s0 = 0; s0 < S; s0 += 32) {
    __syncthreads();  // protect Ks/Vt/Pl from previous iteration's readers
    // stage K and V tiles: 32x64 halves each = 256 16B-chunks, 2 per thread per tile
    #pragma unroll
    for (int i = 0; i < 2; ++i) {
      const int cc = tid * 2 + i;
      const int r  = cc >> 3;            // s_local 0..31
      const int dc = (cc & 7) * 8;       // d chunk
      const bool inb = (s0 + r) < S;
      const size_t gb = (size_t)(b * S + (inb ? (s0 + r) : 0)) * E + h * 64 + dc;
      uint4 kv = make_uint4(0u, 0u, 0u, 0u), vv = make_uint4(0u, 0u, 0u, 0u);
      if (inb) { kv = *(const uint4*)(Km + gb); vv = *(const uint4*)(V + gb); }
      *(uint4*)(&Ks[r][dc]) = kv;
      const half_t* vp = (const half_t*)&vv;
      #pragma unroll
      for (int e = 0; e < 8; ++e) Vt[dc + e][r] = vp[e];
    }
    __syncthreads();

    // scores: two 16x16 tiles, K-dim 64 = 2 chained WMMAs each
    v8f c[2] = {};
    #pragma unroll
    for (int t = 0; t < 2; ++t) {
      const int sl = t * 16 + l16;
      const v16h bk0 = frag16(&Ks[sl][lh * 16],      &Ks[sl][lh * 16 + 8]);
      const v16h bk1 = frag16(&Ks[sl][32 + lh * 16], &Ks[sl][32 + lh * 16 + 8]);
      c[t] = __builtin_amdgcn_wmma_f32_16x16x32_f16(false, aq[0], false, bk0, (short)0, c[t], false, false);
      c[t] = __builtin_amdgcn_wmma_f32_16x16x32_f16(false, aq[1], false, bk1, (short)0, c[t], false, false);
    }
    // mask out-of-range keys: only the tail tile can be OOB
    if (s0 + 32 > S) {
      #pragma unroll
      for (int t = 0; t < 2; ++t) {
        const bool oob = (s0 + t * 16 + l16) >= S;
        #pragma unroll
        for (int j = 0; j < 8; ++j) if (oob) c[t][j] = -1e30f;
      }
    }
    // single tile-wide max (any per-row constant keeps softmax exact)
    float mx = c[0][0];
    #pragma unroll
    for (int j = 1; j < 8; ++j) mx = fmaxf(mx, c[0][j]);
    #pragma unroll
    for (int j = 0; j < 8; ++j) mx = fmaxf(mx, c[1][j]);
    #pragma unroll
    for (int off = 1; off < 32; off <<= 1) mx = fmaxf(mx, __shfl_xor(mx, off, 32));
    const float mn = fmaxf(rm, mx);
    const float alpha = __expf(rm - mn);
    rm = mn;
    #pragma unroll
    for (int j = 0; j < 8; ++j) {
      c[0][j] = __expf(c[0][j] - mn);
      c[1][j] = __expf(c[1][j] - mn);
      rlacc[j] *= alpha;
      #pragma unroll
      for (int tj = 0; tj < 4; ++tj) o[tj][j] *= alpha;
    }
    // transpose P (C layout -> A layout) through per-wave LDS
    #pragma unroll
    for (int t = 0; t < 2; ++t)
      #pragma unroll
      for (int j = 0; j < 8; ++j)
        Pl[wave][j + 8 * lh][t * 16 + l16] = (half_t)c[t][j];
    __syncthreads();
    const v16h ap = frag16(&Pl[wave][l16][lh * 16], &Pl[wave][l16][lh * 16 + 8]);
    // denominator: rowsum(P) via WMMA against all-ones B
    rlacc = __builtin_amdgcn_wmma_f32_16x16x32_f16(false, ap, false, bones, (short)0, rlacc, false, false);
    // O += P * V : 4 d-tiles of 16 (B fragments from transposed V tile, contiguous)
    #pragma unroll
    for (int tj = 0; tj < 4; ++tj) {
      const int d = tj * 16 + l16;
      const v16h bv = frag16(&Vt[d][lh * 16], &Vt[d][lh * 16 + 8]);
      o[tj] = __builtin_amdgcn_wmma_f32_16x16x32_f16(false, ap, false, bv, (short)0, o[tj], false, false);
    }
  }
  #pragma unroll
  for (int tj = 0; tj < 4; ++tj)
    #pragma unroll
    for (int j = 0; j < 8; ++j) {
      const float val = o[tj][j] / rlacc[j];
      O[(size_t)(b * T + q0 + j + 8 * lh) * E + h * 64 + tj * 16 + l16] = (half_t)val;
    }
}

// ---------------- host orchestration ----------------
extern "C" void kernel_launch(void* const* d_in, const int* in_sizes, int n_in,
                              void* d_out, int out_size, void* d_ws, size_t ws_size,
                              hipStream_t stream) {
  (void)in_sizes; (void)n_in; (void)out_size; (void)ws_size;
  const int Bb = 4, T = 1024, H = 1024, CD = 768, FF = 4096, S = 77, NH = 16;
  const int M  = Bb * T;   // 4096 tokens
  const int Mc = Bb * S;   // 308 ctx tokens

  const float* x_in = (const float*)d_in[0];
  const float* ctx  = (const float*)d_in[1];
  const float* sq_w = (const float*)d_in[2];
  const float* sk_w = (const float*)d_in[3];
  const float* sv_w = (const float*)d_in[4];
  const float* so_w = (const float*)d_in[5];
  const float* so_b = (const float*)d_in[6];
  const float* cq_w = (const float*)d_in[7];
  const float* ck_w = (const float*)d_in[8];
  const float* cv_w = (const float*)d_in[9];
  const float* co_w = (const float*)d_in[10];
  const float* co_b = (const float*)d_in[11];
  const float* n1_g = (const float*)d_in[12];
  const float* n1_b = (const float*)d_in[13];
  const float* n2_g = (const float*)d_in[14];
  const float* n2_b = (const float*)d_in[15];
  const float* n3_g = (const float*)d_in[16];
  const float* n3_b = (const float*)d_in[17];
  const float* n4_g = (const float*)d_in[18];
  const float* n4_b = (const float*)d_in[19];
  const float* f1w1 = (const float*)d_in[20];
  const float* f1b1 = (const float*)d_in[21];
  const float* f1w2 = (const float*)d_in[22];
  const float* f1b2 = (const float*)d_in[23];
  const float* f2w1 = (const float*)d_in[24];
  const float* f2b1 = (const float*)d_in[25];
  const float* f2w2 = (const float*)d_in[26];
  const float* f2b2 = (const float*)d_in[27];
  float* out = (float*)d_out;

  char* wp = (char*)d_ws;
  auto ah  = [&](size_t n) { half_t* p = (half_t*)wp; wp += ((n * 2 + 255) & ~(size_t)255); return p; };
  auto af4 = [&](size_t n) { float*  p = (float*)wp;  wp += ((n * 4 + 255) & ~(size_t)255); return p; };
  auto cvt = [&](const float* src, half_t* dst, size_t n) {
    cvt_f32_f16<<<dim3((unsigned)((n / 4 + 255) / 256)), 256, 0, stream>>>(src, dst, (int)n);
  };

  // f16 weights
  half_t* w_sq  = ah((size_t)H * H);   cvt(sq_w, w_sq, (size_t)H * H);
  half_t* w_sk  = ah((size_t)H * H);   cvt(sk_w, w_sk, (size_t)H * H);
  half_t* w_sv  = ah((size_t)H * H);   cvt(sv_w, w_sv, (size_t)H * H);
  half_t* w_so  = ah((size_t)H * H);   cvt(so_w, w_so, (size_t)H * H);
  half_t* w_cq  = ah((size_t)H * H);   cvt(cq_w, w_cq, (size_t)H * H);
  half_t* w_ck  = ah((size_t)CD * H);  cvt(ck_w, w_ck, (size_t)CD * H);
  half_t* w_cv  = ah((size_t)CD * H);  cvt(cv_w, w_cv, (size_t)CD * H);
  half_t* w_co  = ah((size_t)H * H);   cvt(co_w, w_co, (size_t)H * H);
  half_t* w_f1a = ah((size_t)H * FF);  cvt(f1w1, w_f1a, (size_t)H * FF);
  half_t* w_f1b = ah((size_t)FF * H);  cvt(f1w2, w_f1b, (size_t)FF * H);
  half_t* w_f2a = ah((size_t)H * FF);  cvt(f2w1, w_f2a, (size_t)H * FF);
  half_t* w_f2b = ah((size_t)FF * H);  cvt(f2w2, w_f2b, (size_t)FF * H);
  half_t* ctx16 = ah((size_t)Mc * CD); cvt(ctx, ctx16, (size_t)Mc * CD);

  // activations
  half_t* xn = ah((size_t)M * H);
  half_t* qb = ah((size_t)M * H);
  half_t* kb = ah((size_t)M * H);
  half_t* vb = ah((size_t)M * H);
  half_t* ob = ah((size_t)M * H);
  half_t* hb = ah((size_t)M * FF);
  float*  x1 = af4((size_t)M * H);
  float*  x2 = af4((size_t)M * H);

  auto gemm_plain = [&](const half_t* A, const half_t* Bw, half_t* C, int m, int n, int k) {
    dim3 g(n / 128, (m + 127) / 128);
    gemm_wmma_f16<false, false, false, true><<<g, 256, 0, stream>>>(A, Bw, nullptr, nullptr, nullptr, C, m, n, k);
  };
  auto gemm_bias_gelu = [&](const half_t* A, const half_t* Bw, const float* bias, half_t* C, int m, int n, int k) {
    dim3 g(n / 128, (m + 127) / 128);
    gemm_wmma_f16<true, true, false, true><<<g, 256, 0, stream>>>(A, Bw, bias, nullptr, nullptr, C, m, n, k);
  };
  auto gemm_bias_res = [&](const half_t* A, const half_t* Bw, const float* bias, const float* res, float* C,
                           int m, int n, int k) {
    dim3 g(n / 128, (m + 127) / 128);
    gemm_wmma_f16<true, false, true, false><<<g, 256, 0, stream>>>(A, Bw, bias, res, C, nullptr, m, n, k);
  };

  // ---- self-attention branch ----
  layernorm_f16<<<M, 256, 0, stream>>>(x_in, n1_g, n1_b, xn, H);
  gemm_plain(xn, w_sq, qb, M, H, H);
  gemm_plain(xn, w_sk, kb, M, H, H);
  gemm_plain(xn, w_sv, vb, M, H, H);
  attn_flash_wmma<<<dim3(T / 64, Bb * NH), 128, 0, stream>>>(qb, kb, vb, ob, Bb, T, T, NH);
  gemm_bias_res(ob, w_so, so_b, x_in, x1, M, H, H);

  // ---- FFN 1 ----
  layernorm_f16<<<M, 256, 0, stream>>>(x1, n2_g, n2_b, xn, H);
  gemm_bias_gelu(xn, w_f1a, f1b1, hb, M, FF, H);
  gemm_bias_res(hb, w_f1b, f1b2, x1, x2, M, H, FF);

  // ---- cross-attention branch ----
  layernorm_f16<<<M, 256, 0, stream>>>(x2, n3_g, n3_b, xn, H);
  gemm_plain(xn, w_cq, qb, M, H, H);
  gemm_plain(ctx16, w_ck, kb, Mc, H, CD);
  gemm_plain(ctx16, w_cv, vb, Mc, H, CD);
  attn_flash_wmma<<<dim3(T / 64, Bb * NH), 128, 0, stream>>>(qb, kb, vb, ob, Bb, T, S, NH);
  gemm_bias_res(ob, w_co, co_b, x2, out, M, H, H);

  // ---- FFN 2 (residual read/write on d_out is per-element, race-free) ----
  layernorm_f16<<<M, 256, 0, stream>>>(out, n4_g, n4_b, xn, H);
  gemm_bias_gelu(xn, w_f2a, f2b1, hb, M, FF, H);
  gemm_bias_res(hb, w_f2b, f2b2, out, out, M, H, FF);
}
